// MultiHeadPGATLayer_10093173145795
// MI455X (gfx1250) — matrix operand order repaired and checked by reading, not compiled, not run on hardware
//
#include <hip/hip_runtime.h>
#include <math.h>

#define DIN   256
#define DOUT  64
#define NHEAD 4
#define NC    256           // NHEAD * DOUT
#define SLOPE 0.01f
#define NEGV  -1000.0f

typedef __attribute__((ext_vector_type(16))) __bf16 v16bf;
typedef __attribute__((ext_vector_type(8)))  float  v8f;

// ---------- helpers ----------
__device__ __forceinline__ unsigned short f2bf(float f) {
  // round-to-nearest-even f32 -> bf16
  unsigned u = __float_as_uint(f);
  unsigned r = u + 0x7FFFu + ((u >> 16) & 1u);
  return (unsigned short)(r >> 16);
}
__device__ __forceinline__ unsigned enc_f32(float f) {
  // monotone float -> uint map for atomicMax
  unsigned u = __float_as_uint(f);
  return (u & 0x80000000u) ? ~u : (u | 0x80000000u);
}
__device__ __forceinline__ float dec_f32(unsigned k) {
  unsigned u = (k & 0x80000000u) ? (k & 0x7FFFFFFFu) : ~k;
  return __uint_as_float(u);
}

// ---------- 0: init out / denom / segment-max keys ----------
__global__ void init_kernel(float* __restrict__ out, long long outN,
                            float* __restrict__ denom, unsigned* __restrict__ mkey,
                            int nh) {
  long long i = (long long)blockIdx.x * blockDim.x + threadIdx.x;
  if (i < outN) out[i] = 0.0f;
  if (i < nh) { denom[i] = 0.0f; mkey[i] = 0x007FFFFFu; }   // enc(-inf)
}

// ---------- 1: pack h -> bf16 A-fragments (16x32 per (tile,kstep)) ----------
// A layout (16-bit A 16x32, ISA 7.12.2): lane<16: row=lane,  K in {0..7, 16..23}
//                                        lane>=16: row=lane-16, K in {8..15, 24..31}
__global__ void pack_a(const float* __restrict__ h, unsigned short* __restrict__ A, int N) {
  int tile = blockIdx.x;
  int s    = threadIdx.x >> 5;        // k-step 0..7
  int lane = threadIdx.x & 31;
  int row  = tile * 16 + (lane & 15);
  int kb   = (lane < 16) ? 0 : 8;
  unsigned short* o = A + (((size_t)tile * 8 + s) * 32 + lane) * 16;
  if (row < N) {
    const float* hr = h + (size_t)row * DIN + s * 32 + kb;
#pragma unroll
    for (int j = 0; j < 8; ++j) o[j]     = f2bf(hr[j]);
#pragma unroll
    for (int j = 0; j < 8; ++j) o[8 + j] = f2bf(hr[16 + j]);
  } else {
#pragma unroll
    for (int j = 0; j < 16; ++j) o[j] = 0;
  }
}

// ---------- 2: pack W -> bf16 B-fragments (32x16 per (coltile,kstep)) ----------
// B[d][c] = W[c/64][d][c%64].  B layout: lane<16: col=lane, K in {0..7, 16..23}; lane>=16: col=lane-16, K in {8..15, 24..31}
__global__ void pack_b(const float* __restrict__ W, unsigned short* __restrict__ B) {
  int ct   = blockIdx.x;              // 0..15
  int s    = threadIdx.x >> 5;        // 0..7
  int lane = threadIdx.x & 31;
  int c    = ct * 16 + (lane & 15);
  int head = c >> 6, f = c & 63;
  int kbase = s * 32 + ((lane < 16) ? 0 : 8);
  unsigned short* o = B + (((size_t)ct * 8 + s) * 32 + lane) * 16;
#pragma unroll
  for (int j = 0; j < 16; ++j)
    o[j] = f2bf(W[((size_t)head * DIN + (kbase + j + (j >= 8 ? 8 : 0))) * DOUT + f]);
}

// ---------- 3: z = h @ B  via v_wmma_f32_16x16x32_bf16 ----------
// block = 256 threads = 8 waves. Block covers row-tile blockIdx.x, wave w covers col-tiles 2w, 2w+1.
__global__ void __launch_bounds__(256) gemm_z(const unsigned short* __restrict__ Aws,
                                              const unsigned short* __restrict__ Bws,
                                              float* __restrict__ z, int N) {
  int tile = blockIdx.x;
  int wave = threadIdx.x >> 5;
  int lane = threadIdx.x & 31;
  int ct0 = wave * 2, ct1 = wave * 2 + 1;

  const v16bf* Af = (const v16bf*)Aws + (size_t)tile * 8 * 32 + lane;
  const v16bf* Bf = (const v16bf*)Bws;

  v8f acc0 = {};
  v8f acc1 = {};
#pragma unroll
  for (int s = 0; s < 8; ++s) {
    v16bf a  = Af[(size_t)s * 32];
    v16bf b0 = Bf[((size_t)ct0 * 8 + s) * 32 + lane];
    v16bf b1 = Bf[((size_t)ct1 * 8 + s) * 32 + lane];
    acc0 = __builtin_amdgcn_wmma_f32_16x16x32_bf16(false, a, false, b0, (short)0, acc0, false, false);
    acc1 = __builtin_amdgcn_wmma_f32_16x16x32_bf16(false, a, false, b1, (short)0, acc1, false, false);
  }
  // C/D layout: lane<16 -> M=r, N=lane ; lane>=16 -> M=r+8, N=lane-16
  int m0 = (lane >> 4) * 8;
  int n  = lane & 15;
#pragma unroll
  for (int r = 0; r < 8; ++r) {
    int row = tile * 16 + m0 + r;
    if (row < N) {
      z[(size_t)row * NC + ct0 * 16 + n] = acc0[r];
      z[(size_t)row * NC + ct1 * 16 + n] = acc1[r];
    }
  }
}

// ---------- 4: es/ed per (node, head): wave-level dot with att ----------
__global__ void scores_kernel(const float* __restrict__ z, const float* __restrict__ att,
                              float* __restrict__ es, float* __restrict__ ed, int N) {
  int w    = (int)(((long long)blockIdx.x * blockDim.x + threadIdx.x) >> 5);
  int lane = threadIdx.x & 31;
  if (w >= N * NHEAD) return;
  int n = w >> 2, hh = w & 3;
  const float* zr = z + (size_t)n * NC + hh * DOUT;
  const float* as = att + (size_t)hh * 2 * DOUT;
  const float* ad = as + DOUT;
  int f0 = lane * 2;
  float z0 = zr[f0], z1 = zr[f0 + 1];
  float s = z0 * as[f0] + z1 * as[f0 + 1];
  float d = z0 * ad[f0] + z1 * ad[f0 + 1];
#pragma unroll
  for (int off = 16; off; off >>= 1) {
    s += __shfl_xor(s, off, 32);
    d += __shfl_xor(d, off, 32);
  }
  if (lane == 0) { es[w] = s; ed[w] = d; }
}

// ---------- 5: e = lrelu(es[src]+ed[dst]); 0 -> NEG; segment-max into mkey ----------
__global__ void edge_pass1(const int* __restrict__ src, const int* __restrict__ dst,
                           const float* __restrict__ es, const float* __restrict__ ed,
                           float* __restrict__ e, unsigned* __restrict__ mkey, int E) {
  int i = blockIdx.x * blockDim.x + threadIdx.x;
  if (i >= E) return;
  int s = src[i], d = dst[i];
  float4 vs = *(const float4*)(es + (size_t)s * 4);
  float4 vd = *(const float4*)(ed + (size_t)d * 4);
  float v[4] = { vs.x + vd.x, vs.y + vd.y, vs.z + vd.z, vs.w + vd.w };
  float4 eo;
  float* eop = &eo.x;
#pragma unroll
  for (int hh = 0; hh < 4; ++hh) {
    float x = v[hh];
    x = (x > 0.0f) ? x : SLOPE * x;
    if (x == 0.0f) x = NEGV;
    eop[hh] = x;
    atomicMax(&mkey[(size_t)d * 4 + hh], enc_f32(x));
  }
  *(float4*)(e + (size_t)i * 4) = eo;
}

// ---------- 6: m = finite(dec(mkey)) ? . : 0 ----------
__global__ void node_m(const unsigned* __restrict__ mkey, float* __restrict__ m, int nh) {
  int i = blockIdx.x * blockDim.x + threadIdx.x;
  if (i >= nh) return;
  float v = dec_f32(mkey[i]);
  unsigned u = __float_as_uint(v);
  bool fin = ((u >> 23) & 0xFFu) != 0xFFu;
  m[i] = fin ? v : 0.0f;
}

// ---------- 7: ex = exp(e - m[dst]); denom += ex ----------
__global__ void edge_pass2(const int* __restrict__ dst, const float* __restrict__ e,
                           const float* __restrict__ m, float* __restrict__ ex,
                           float* __restrict__ denom, int E) {
  int i = blockIdx.x * blockDim.x + threadIdx.x;
  if (i >= E) return;
  int d = dst[i];
  float4 ev = *(const float4*)(e + (size_t)i * 4);
  float4 mv = *(const float4*)(m + (size_t)d * 4);
  float4 xo;
  xo.x = __expf(ev.x - mv.x);
  xo.y = __expf(ev.y - mv.y);
  xo.z = __expf(ev.z - mv.z);
  xo.w = __expf(ev.w - mv.w);
  *(float4*)(ex + (size_t)i * 4) = xo;
  atomicAdd(&denom[(size_t)d * 4 + 0], xo.x);
  atomicAdd(&denom[(size_t)d * 4 + 1], xo.y);
  atomicAdd(&denom[(size_t)d * 4 + 2], xo.z);
  atomicAdd(&denom[(size_t)d * 4 + 3], xo.w);
}

// ---------- 8: out[dst] += alpha * z[src], one wave32 per edge ----------
// EPW edges per wave (a block of consecutive edges per wave) so we can software-prefetch
// the next edge's z row (global_prefetch_b8) while the current edge's gather
// and atomics are in flight — hides L2 latency in the dominant phase.
#define EPW 4
__global__ void __launch_bounds__(256) edge_pass3(const int* __restrict__ src, const int* __restrict__ dst,
                                                  const float* __restrict__ ex, const float* __restrict__ denom,
                                                  const float* __restrict__ z, float* __restrict__ out, int E) {
  int w    = (int)(((long long)blockIdx.x * blockDim.x + threadIdx.x) >> 5);
  int lane = threadIdx.x & 31;
  int base = w * EPW;
  if (base >= E) return;

  // prefetch the first edge's z row (lanes 32B apart cover the 1KB row)
  {
    int s0 = src[base];
    __builtin_prefetch(z + (size_t)s0 * NC + lane * 8, 0, 1);
  }

#pragma unroll
  for (int k = 0; k < EPW; ++k) {
    int i = base + k;
    if (i >= E) break;

    // prefetch next edge's z row before touching this edge's data
    if (k + 1 < EPW && i + 1 < E) {
      int sn = src[i + 1];
      __builtin_prefetch(z + (size_t)sn * NC + lane * 8, 0, 1);
    }

    int s = src[i], d = dst[i];
    float4 exv = *(const float4*)(ex + (size_t)i * 4);
    float4 dn  = *(const float4*)(denom + (size_t)d * 4);
    float coef[4];
    coef[0] = exv.x / ((dn.x > 0.0f) ? dn.x : 1.0f);
    coef[1] = exv.y / ((dn.y > 0.0f) ? dn.y : 1.0f);
    coef[2] = exv.z / ((dn.z > 0.0f) ? dn.z : 1.0f);
    coef[3] = exv.w / ((dn.w > 0.0f) ? dn.w : 1.0f);
    const float* zr   = z   + (size_t)s * NC;
    float*       orow = out + (size_t)d * NC;
#pragma unroll
    for (int j = 0; j < 8; ++j) {
      int c = lane + 32 * j;             // c>>6 == j>>1 for all lanes -> constant head index
      atomicAdd(&orow[c], coef[j >> 1] * zr[c]);
    }
  }
}

// ---------- launch ----------
extern "C" void kernel_launch(void* const* d_in, const int* in_sizes, int n_in,
                              void* d_out, int out_size, void* d_ws, size_t ws_size,
                              hipStream_t stream) {
  const float* h   = (const float*)d_in[0];
  const float* W   = (const float*)d_in[1];
  const float* att = (const float*)d_in[2];
  const int*   src = (const int*)d_in[3];
  const int*   dst = (const int*)d_in[4];
  float* out = (float*)d_out;

  int N = in_sizes[0] / DIN;
  int E = in_sizes[3];
  int ntiles = (N + 15) / 16;

  char* ws = (char*)d_ws;
  size_t off = 0;
  auto carve = [&](size_t bytes) -> void* {
    void* p = ws + off;
    off = (off + bytes + 255) & ~(size_t)255;
    return p;
  };
  unsigned short* Aws  = (unsigned short*)carve((size_t)ntiles * 8 * 32 * 16 * 2); // 51.2 MB
  unsigned short* Bws  = (unsigned short*)carve((size_t)16 * 8 * 32 * 16 * 2);     // 128 KB
  float*          z    = (float*)carve((size_t)N * NC * 4);                        // 102.4 MB
  float*          es   = (float*)carve((size_t)N * NHEAD * 4);
  float*          ed   = (float*)carve((size_t)N * NHEAD * 4);
  float*          ebuf = (float*)carve((size_t)E * NHEAD * 4);                     // 25.6 MB
  float*          exb  = (float*)carve((size_t)E * NHEAD * 4);                     // 25.6 MB
  unsigned*       mkey = (unsigned*)carve((size_t)N * NHEAD * 4);
  float*          mval = (float*)carve((size_t)N * NHEAD * 4);
  float*          dnm  = (float*)carve((size_t)N * NHEAD * 4);

  long long outN = (long long)N * NC;
  int nh = N * NHEAD;

  init_kernel<<<(unsigned)((outN + 255) / 256), 256, 0, stream>>>(out, outN, dnm, mkey, nh);
  pack_a<<<ntiles, 256, 0, stream>>>(h, Aws, N);
  pack_b<<<16, 256, 0, stream>>>(W, Bws);
  gemm_z<<<ntiles, 256, 0, stream>>>(Aws, Bws, z, N);
  scores_kernel<<<(nh * 32 + 255) / 256, 256, 0, stream>>>(z, att, es, ed, N);
  edge_pass1<<<(E + 255) / 256, 256, 0, stream>>>(src, dst, es, ed, ebuf, mkey, E);
  node_m<<<(nh + 255) / 256, 256, 0, stream>>>(mkey, mval, nh);
  edge_pass2<<<(E + 255) / 256, 256, 0, stream>>>(dst, ebuf, mval, exb, dnm, E);

  int waves3  = (E + EPW - 1) / EPW;
  int blocks3 = (waves3 + 7) / 8;
  edge_pass3<<<blocks3, 256, 0, stream>>>(src, dst, exb, dnm, z, out, E);
}
